// MRNAtten_50337016709507
// MI455X (gfx1250) — compile-verified, matrix operand order checked by reading, not run on hardware
//
#include <hip/hip_runtime.h>
#include <hip/hip_bf16.h>
#include <math.h>

// ---------------------------------------------------------------------------
// Problem constants: N=4096, B=4, NPM=1024, E=16384, H=512, HEADS=6, LOOPS=3
// ---------------------------------------------------------------------------
constexpr int   NN    = 4096;
constexpr int   BB    = 4;
constexpr int   NPMc  = 1024;
constexpr int   EEDGE = 16384;
constexpr int   HH    = 512;
constexpr int   NHEAD = 6;
constexpr int   NLOOP = 3;
constexpr int   EE    = EEDGE + NN;          // 20480 (edges + self loops)
constexpr int   FDIM  = HH + 2;              // 514 real feat width
constexpr int   FPAD  = 576;                 // padded to multiple of 64
constexpr int   GATW  = NHEAD * HH;          // 3072

// Workspace layout (floats). xl overlays (qkv+scores+o): 6,291,456 +
// 4,194,304 + 2,097,152 == 12,582,912 == N*HEADS*H exactly.
constexpr long long SZ_H    = (long long)NN * HH;
constexpr long long SZ_HN   = (long long)NN * FPAD;        // ln out / padded feat
constexpr long long SZ_QKV  = (long long)NN * 3 * HH;
constexpr long long SZ_SC   = (long long)BB * NPMc * NPMc;
constexpr long long SZ_O    = (long long)NN * HH;
constexpr long long SZ_XG   = (long long)NN * GATW;
constexpr long long SZ_LOG  = (long long)EE * NHEAD;
constexpr long long SZ_MZ   = (long long)NN * NHEAD;

constexpr long long OFF_H    = 0;
constexpr long long OFF_HN   = OFF_H   + SZ_H;
constexpr long long OFF_QKV  = OFF_HN  + SZ_HN;
constexpr long long OFF_SC   = OFF_QKV + SZ_QKV;
constexpr long long OFF_O    = OFF_SC  + SZ_SC;
constexpr long long OFF_XL   = OFF_QKV;                    // overlays qkv/sc/o
constexpr long long OFF_XR   = OFF_XL  + SZ_XG;
constexpr long long OFF_GOUT = OFF_XR  + SZ_XG;
constexpr long long OFF_LOG  = OFF_GOUT + SZ_XG;
constexpr long long OFF_EA   = OFF_LOG + SZ_LOG;
constexpr long long OFF_MX   = OFF_EA  + SZ_LOG;
constexpr long long OFF_Z    = OFF_MX  + SZ_MZ;
constexpr long long OFF_CO   = OFF_Z   + SZ_MZ;            // coord: 2*N

// ---------------------------------------------------------------------------
// WMMA types (gfx1250, wave32)
// ---------------------------------------------------------------------------
typedef __attribute__((ext_vector_type(16))) _Float16 v16h;
typedef __attribute__((ext_vector_type(8)))  float    v8f;

__device__ __forceinline__ float selu_f(float x) {
  const float sc = 1.0507009873554805f, al = 1.6732632423543772f;
  return x > 0.f ? sc * x : sc * al * (expf(x) - 1.f);
}
__device__ __forceinline__ float gelu_f(float x) {
  return 0.5f * x * (1.f + erff(x * 0.70710678118654752f));
}
__device__ __forceinline__ float atomicMaxFloat(float* addr, float val) {
  if (val >= 0.f)
    return __int_as_float(atomicMax((int*)addr, __float_as_int(val)));
  return __uint_as_float(atomicMin((unsigned int*)addr, __float_as_uint(val)));
}
__device__ __forceinline__ void edge_sd(const int* __restrict__ ei, int e,
                                        int& s, int& d) {
  if (e < EEDGE) { s = ei[e]; d = ei[EEDGE + e]; }
  else           { s = e - EEDGE; d = e - EEDGE; }
}

// ---------------------------------------------------------------------------
// Batched WMMA GEMM:  C = act(scale * A*B + bias) + resid
//   Block = 128 threads (4 waves): 32(M) x 64(N) output slab.  Each wave owns
//   one 16(N)-tile and TWO 16(M)-tiles that share the staged B tile (halves
//   L2 B-traffic, 4 WMMAs per barrier window).  K stepped by 64 via LDS.
//   Caller guarantees M%32==0, Nn%64==0, K%64==0.
//   GUARDB (template): B has only Kreal (<K) valid K rows/cols, handled
//   branchlessly (clamped addresses + zero masks -> v_min/v_cndmask).
//   TRANSB (template): B is (Nn x K) row-major when 1, (K x Nn) when 0.
//   A fragment layout (ISA 7.12.2, 16-bit A 16x32): lane L -> row = L%16,
//   half = L/16; VGPR v<4 -> K = 2v+8*half (+0/1); v>=4 -> K = 16+2(v-4)+8*half.
//   B mirrors with N = L%16.  C/D: VGPR r -> row = 8*half + r.
// ---------------------------------------------------------------------------
#define GTM 32  // block tile M (2 x 16 sub-tiles per wave)
#define GTN 64  // block tile N
#define GTK 64  // K step per barrier window (2 WMMA sub-steps of 32)
#define LSTR 68 // LDS row stride (floats): 272B rows, 16B-aligned, bank-spread

template <int GUARDB, int TRANSB>
__global__ __launch_bounds__(128)
void k_gemm_wmma(const float* __restrict__ A, int lda, long long sA,
                 const float* __restrict__ Bm, int ldb, long long sB,
                 const float* __restrict__ bias,
                 const float* __restrict__ resid,
                 float* __restrict__ C, int ldc, long long sC,
                 int M, int Nn, int K, int Kreal, float scale, int act) {
  __shared__ float lds_a[GTM][LSTR];    // A tile: 32 x 64
  __shared__ float lds_bt[GTN][LSTR];   // B^T tile: 64(n) x 64(k)

  const int tid  = threadIdx.x;
  const int lane = tid & 31;
  const int wave = tid >> 5;
  const int m0   = blockIdx.y * GTM;
  const int n0   = blockIdx.x * GTN;
  const float* Ab = A  + (long long)blockIdx.z * sA;
  const float* Bb = Bm + (long long)blockIdx.z * sB;

  const int row16 = lane & 15;
  const int half  = lane >> 4;

  v8f acc0 = {}, acc1 = {};
  for (int k0 = 0; k0 < K; k0 += GTK) {
    // ---- stage A tile: 32 rows x 64 cols (4 float4 per thread) ----
    {
      const int ar = tid >> 2;              // 0..31
      const int ac = (tid & 3) * 16;        // 0,16,32,48
      const float* pa = &Ab[(size_t)(m0 + ar) * lda + k0 + ac];
#pragma unroll
      for (int j = 0; j < 4; ++j) {
        float4 v = *(const float4*)(pa + j * 4);
        *(float4*)&lds_a[ar][ac + j * 4] = v;
      }
    }
    // ---- stage B^T tile: 64(n) x 64(k), 32 floats per thread ----
    if (TRANSB) {
      // B is Nn x K row-major: row n contiguous in k
      const int nrow = tid >> 1;            // 0..63
      const int kc   = (tid & 1) * 32;      // 0 or 32
#pragma unroll
      for (int j = 0; j < 8; ++j) {
        const int kg = k0 + kc + j * 4;
        float4 v;
        if (GUARDB) {
          const float* pr = &Bb[(size_t)(n0 + nrow) * ldb];
          v.x = (kg + 0 < Kreal) ? pr[(kg + 0 < Kreal) ? kg + 0 : 0] : 0.f;
          v.y = (kg + 1 < Kreal) ? pr[(kg + 1 < Kreal) ? kg + 1 : 0] : 0.f;
          v.z = (kg + 2 < Kreal) ? pr[(kg + 2 < Kreal) ? kg + 2 : 0] : 0.f;
          v.w = (kg + 3 < Kreal) ? pr[(kg + 3 < Kreal) ? kg + 3 : 0] : 0.f;
        } else {
          v = *(const float4*)&Bb[(size_t)(n0 + nrow) * ldb + kg];
        }
        *(float4*)&lds_bt[nrow][kc + j * 4] = v;
      }
    } else {
      // B is K x Nn row-major: row k contiguous in n -> transpose via LDS
      const int krow = tid >> 1;            // 0..63
      const int nc   = (tid & 1) * 32;      // 0 or 32
      const int kg   = k0 + krow;
      float msk = 1.f;
      int   krd = kg;
      if (GUARDB) {
        msk = (kg < Kreal) ? 1.f : 0.f;
        krd = (kg < Kreal) ? kg : (Kreal - 1);
      }
      const float* pr = &Bb[(size_t)krd * ldb + n0 + nc];
#pragma unroll
      for (int j = 0; j < 8; ++j) {
        float4 v = *(const float4*)(pr + j * 4);
        lds_bt[nc + j * 4 + 0][krow] = v.x * msk;
        lds_bt[nc + j * 4 + 1][krow] = v.y * msk;
        lds_bt[nc + j * 4 + 2][krow] = v.z * msk;
        lds_bt[nc + j * 4 + 3][krow] = v.w * msk;
      }
    }
    __syncthreads();

    // ---- 2 K-substeps x 2 M-subtiles = 4 WMMAs per barrier window ----
#pragma unroll
    for (int s = 0; s < 2; ++s) {
      v16h afrag0, afrag1, bfrag;
#pragma unroll
      for (int v = 0; v < 8; ++v) {
        const int kb = s * 32 +
            ((v < 4) ? (v * 2 + half * 8) : (16 + (v - 4) * 2 + half * 8));
        const float2 a0 = *(const float2*)&lds_a[row16][kb];
        const float2 a1 = *(const float2*)&lds_a[16 + row16][kb];
        const float2 b2 = *(const float2*)&lds_bt[wave * 16 + row16][kb];
        afrag0[v * 2 + 0] = (_Float16)a0.x;
        afrag0[v * 2 + 1] = (_Float16)a0.y;
        afrag1[v * 2 + 0] = (_Float16)a1.x;
        afrag1[v * 2 + 1] = (_Float16)a1.y;
        bfrag[v * 2 + 0]  = (_Float16)b2.x;
        bfrag[v * 2 + 1]  = (_Float16)b2.y;
      }
      acc0 = __builtin_amdgcn_wmma_f32_16x16x32_f16(
          false, afrag0, false, bfrag, (short)0, acc0, false, false);
      acc1 = __builtin_amdgcn_wmma_f32_16x16x32_f16(
          false, afrag1, false, bfrag, (short)0, acc1, false, false);
    }
    __syncthreads();
  }

  // ---- epilogue (M, Nn exact multiples of tile sizes: unconditional) ----
  const int n = n0 + wave * 16 + row16;
  const float bval = bias ? bias[n] : 0.f;
  float* Cb = C + (long long)blockIdx.z * sC;
  const float* Rb = resid ? (resid + (long long)blockIdx.z * sC) : nullptr;
#pragma unroll
  for (int r = 0; r < 8; ++r) {
    const int row0 = m0 + half * 8 + r;
    const int row1 = row0 + 16;
    float v0 = acc0[r] * scale + bval;
    float v1 = acc1[r] * scale + bval;
    if (act == 1) { v0 = gelu_f(v0); v1 = gelu_f(v1); }
    if (Rb) {
      v0 += Rb[(size_t)row0 * ldc + n];
      v1 += Rb[(size_t)row1 * ldc + n];
    }
    Cb[(size_t)row0 * ldc + n] = v0;
    Cb[(size_t)row1 * ldc + n] = v1;
  }
}

// ---------------------------------------------------------------------------
// Input linear: h = selu([x[:,2:7] | local_feat | conv_feat[n/NPM]] @ W + b)
// ---------------------------------------------------------------------------
__global__ void k_lin_selu(const float* __restrict__ x,
                           const float* __restrict__ lf,
                           const float* __restrict__ cf,
                           const float* __restrict__ W,
                           const float* __restrict__ b,
                           float* __restrict__ h) {
  const int idx = blockIdx.x * blockDim.x + threadIdx.x;
  if (idx >= NN * HH) return;
  const int n = idx >> 9, j = idx & (HH - 1);
  float acc = b[j];
#pragma unroll
  for (int k = 0; k < 5; ++k)  acc += x[n * 7 + 2 + k] * W[k * HH + j];
#pragma unroll
  for (int k = 0; k < 16; ++k) acc += lf[n * 16 + k] * W[(5 + k) * HH + j];
  const int bb = n >> 10;  // node_num is uniformly NPM=1024
#pragma unroll
  for (int k = 0; k < 16; ++k) acc += cf[bb * 16 + k] * W[(21 + k) * HH + j];
  h[idx] = selu_f(acc);
}

__global__ void k_init_coord(const float* __restrict__ x, float* __restrict__ c) {
  const int n = blockIdx.x * blockDim.x + threadIdx.x;
  if (n >= NN) return;
  c[2 * n]     = x[n * 7];
  c[2 * n + 1] = x[n * 7 + 1];
}

// ---------------------------------------------------------------------------
// Row LayerNorm over width 512, optional residual: Y = ln(X)*s + b (+ R)
// ---------------------------------------------------------------------------
__global__ void k_layernorm(const float* __restrict__ X,
                            const float* __restrict__ sc,
                            const float* __restrict__ bi,
                            const float* __restrict__ resid,
                            float* __restrict__ Y) {
  const int row = blockIdx.x;
  const float* p = X + (size_t)row * HH;
  __shared__ float s1[256], s2[256];
  float a = 0.f, b = 0.f;
  for (int i = threadIdx.x; i < HH; i += 256) { float v = p[i]; a += v; b += v * v; }
  s1[threadIdx.x] = a; s2[threadIdx.x] = b; __syncthreads();
  for (int s = 128; s > 0; s >>= 1) {
    if (threadIdx.x < s) { s1[threadIdx.x] += s1[threadIdx.x + s];
                           s2[threadIdx.x] += s2[threadIdx.x + s]; }
    __syncthreads();
  }
  const float mu   = s1[0] * (1.f / HH);
  const float var  = s2[0] * (1.f / HH) - mu * mu;
  const float rstd = rsqrtf(var + 1e-5f);
  for (int i = threadIdx.x; i < HH; i += 256) {
    float v = (p[i] - mu) * rstd * sc[i] + bi[i];
    if (resid) v += resid[(size_t)row * HH + i];
    Y[(size_t)row * HH + i] = v;
  }
}

// Row softmax, width 1024 (scores contiguous 4096 x 1024)
__global__ void k_softmax(float* __restrict__ S, int W) {
  const int row = blockIdx.x;
  float* p = S + (size_t)row * W;
  __shared__ float sh[256];
  float mx = -INFINITY;
  for (int i = threadIdx.x; i < W; i += 256) mx = fmaxf(mx, p[i]);
  sh[threadIdx.x] = mx; __syncthreads();
  for (int s = 128; s > 0; s >>= 1) {
    if (threadIdx.x < s) sh[threadIdx.x] = fmaxf(sh[threadIdx.x], sh[threadIdx.x + s]);
    __syncthreads();
  }
  mx = sh[0]; __syncthreads();
  float sum = 0.f;
  for (int i = threadIdx.x; i < W; i += 256) { float e = expf(p[i] - mx); p[i] = e; sum += e; }
  sh[threadIdx.x] = sum; __syncthreads();
  for (int s = 128; s > 0; s >>= 1) {
    if (threadIdx.x < s) sh[threadIdx.x] += sh[threadIdx.x + s];
    __syncthreads();
  }
  const float inv = 1.f / sh[0];
  for (int i = threadIdx.x; i < W; i += 256) p[i] *= inv;
}

// ---------------------------------------------------------------------------
// GAT kernels
// ---------------------------------------------------------------------------
__global__ void k_build_feat(const float* __restrict__ coord,
                             const float* __restrict__ hidden,
                             float* __restrict__ feat) {
  const int idx = blockIdx.x * blockDim.x + threadIdx.x;
  if (idx >= NN * FPAD) return;
  const int n = idx / FPAD, c = idx - n * FPAD;
  float v = 0.f;
  if (c < 2)         v = coord[2 * n + c];
  else if (c < FDIM) v = hidden[(size_t)n * HH + (c - 2)];
  feat[idx] = v;   // cols [514,576) zero-padded for the WMMA K loop
}

__global__ void k_fill(float* __restrict__ p, float v, long long cnt) {
  const long long i = (long long)blockIdx.x * blockDim.x + threadIdx.x;
  if (i < cnt) p[i] = v;
}

// wave per (edge, head): logit = sum_c leaky(xl[src]+xr[dst]) * att[h]
__global__ void k_edge_logits(const int* __restrict__ ei,
                              const float* __restrict__ xl,
                              const float* __restrict__ xr,
                              const float* __restrict__ att,
                              float* __restrict__ logits,
                              float* __restrict__ mx) {
  const int gw   = (blockIdx.x * blockDim.x + threadIdx.x) >> 5;
  const int lane = threadIdx.x & 31;
  if (gw >= EE * NHEAD) return;
  const int e = gw / NHEAD, hh = gw - e * NHEAD;
  int src, dst; edge_sd(ei, e, src, dst);
  const float* pl = xl + (size_t)src * GATW + hh * HH;
  const float* pr = xr + (size_t)dst * GATW + hh * HH;
  const float* pa = att + hh * HH;
  float acc = 0.f;
  for (int c = lane; c < HH; c += 32) {
    float m = pl[c] + pr[c];
    m = m > 0.f ? m : 0.2f * m;
    acc += m * pa[c];
  }
#pragma unroll
  for (int off = 16; off > 0; off >>= 1) acc += __shfl_xor(acc, off, 32);
  if (lane == 0) {
    logits[(size_t)e * NHEAD + hh] = acc;
    atomicMaxFloat(&mx[dst * NHEAD + hh], acc);
  }
}

__global__ void k_edge_exp(const int* __restrict__ ei,
                           const float* __restrict__ logits,
                           const float* __restrict__ mx,
                           float* __restrict__ ea,
                           float* __restrict__ z) {
  const int i = blockIdx.x * blockDim.x + threadIdx.x;
  if (i >= EE * NHEAD) return;
  const int e = i / NHEAD, hh = i - e * NHEAD;
  int src, dst; edge_sd(ei, e, src, dst);
  (void)src;
  const float ev = expf(logits[i] - mx[dst * NHEAD + hh]);
  ea[i] = ev;
  atomicAdd(&z[dst * NHEAD + hh], ev);
}

// block per edge: gout[dst] += alpha[e,h] * xl[src]   (3072 floats / edge)
__global__ void k_edge_scatter(const int* __restrict__ ei,
                               const float* __restrict__ ea,
                               const float* __restrict__ z,
                               const float* __restrict__ xl,
                               float* __restrict__ gout) {
  const int e = blockIdx.x;
  int src, dst; edge_sd(ei, e, src, dst);
  __shared__ float salpha[NHEAD];
  if (threadIdx.x < NHEAD)
    salpha[threadIdx.x] = ea[(size_t)e * NHEAD + threadIdx.x]
                        / (z[dst * NHEAD + threadIdx.x] + 1e-16f);
  __syncthreads();
  for (int i = threadIdx.x; i < GATW; i += blockDim.x) {
    const int hh = i >> 9;
    atomicAdd(&gout[(size_t)dst * GATW + i],
              salpha[hh] * xl[(size_t)src * GATW + i]);
  }
}

__global__ void k_gat_hidden(const float* __restrict__ gout,
                             const float* __restrict__ bias,
                             float* __restrict__ hidden) {
  const int idx = blockIdx.x * blockDim.x + threadIdx.x;
  if (idx >= NN * HH) return;
  const int n = idx >> 9, c = idx & (HH - 1);
  float acc = 0.f;
#pragma unroll
  for (int hh = 0; hh < NHEAD; ++hh)
    acc += gout[(size_t)n * GATW + hh * HH + c];
  hidden[idx] = selu_f(acc * (1.f / NHEAD) + bias[c]);
}

// wave per node: oc = hidden @ coord_w + coord_b, then boundary masking
__global__ void k_coord_update(const float* __restrict__ hidden,
                               const float* __restrict__ cw,
                               const float* __restrict__ cb,
                               float* __restrict__ coord) {
  const int gw   = (blockIdx.x * blockDim.x + threadIdx.x) >> 5;
  const int lane = threadIdx.x & 31;
  if (gw >= NN) return;
  const float c0 = coord[2 * gw], c1 = coord[2 * gw + 1];
  const bool up = (c0 == 1.f), down = (c0 == 0.f);
  const bool left = (c1 == 0.f), right = (c1 == 1.f);
  float a0 = 0.f, a1 = 0.f;
  for (int c = lane; c < HH; c += 32) {
    const float hv = hidden[(size_t)gw * HH + c];
    a0 += hv * cw[c * 2];
    a1 += hv * cw[c * 2 + 1];
  }
#pragma unroll
  for (int off = 16; off > 0; off >>= 1) {
    a0 += __shfl_xor(a0, off, 32);
    a1 += __shfl_xor(a1, off, 32);
  }
  if (lane == 0) {
    const float oc0 = a0 + cb[0], oc1 = a1 + cb[1];
    coord[2 * gw]     = down  ? 0.f : (up   ? 1.f : oc0);
    coord[2 * gw + 1] = right ? 1.f : (left ? 0.f : oc1);
  }
}

__global__ void k_copy(const float* __restrict__ s, float* __restrict__ d, int n) {
  const int i = blockIdx.x * blockDim.x + threadIdx.x;
  if (i < n) d[i] = s[i];
}

// ---------------------------------------------------------------------------
// Host-side GEMM launcher (M%32==0, Nn%64==0, Kpad%64==0 guaranteed by caller)
// ---------------------------------------------------------------------------
static void launch_gemm(hipStream_t st, const float* A, int lda, long long sA,
                        const float* Bm, int ldb, long long sB, bool transB,
                        const float* bias, const float* resid,
                        float* C, int ldc, long long sC,
                        int M, int Nn, int Kpad, int Kreal,
                        float scale, int act, int batches) {
  dim3 grid(Nn / GTN, M / GTM, batches), block(128, 1, 1);
  if (Kpad != Kreal) {
    if (transB)
      k_gemm_wmma<1, 1><<<grid, block, 0, st>>>(A, lda, sA, Bm, ldb, sB,
          bias, resid, C, ldc, sC, M, Nn, Kpad, Kreal, scale, act);
    else
      k_gemm_wmma<1, 0><<<grid, block, 0, st>>>(A, lda, sA, Bm, ldb, sB,
          bias, resid, C, ldc, sC, M, Nn, Kpad, Kreal, scale, act);
  } else {
    if (transB)
      k_gemm_wmma<0, 1><<<grid, block, 0, st>>>(A, lda, sA, Bm, ldb, sB,
          bias, resid, C, ldc, sC, M, Nn, Kpad, Kreal, scale, act);
    else
      k_gemm_wmma<0, 0><<<grid, block, 0, st>>>(A, lda, sA, Bm, ldb, sB,
          bias, resid, C, ldc, sC, M, Nn, Kpad, Kreal, scale, act);
  }
}

extern "C" void kernel_launch(void* const* d_in, const int* in_sizes, int n_in,
                              void* d_out, int out_size, void* d_ws, size_t ws_size,
                              hipStream_t stream) {
  const float* x         = (const float*)d_in[0];
  const float* conv_feat = (const float*)d_in[1];
  const float* local_feat= (const float*)d_in[2];
  // d_in[3] node_num: uniformly NPM per reference setup
  const int*   edge_index= (const int*)d_in[4];
  const float* lin_w     = (const float*)d_in[5];
  const float* lin_b     = (const float*)d_in[6];
  const float* ln1_s     = (const float*)d_in[7];
  const float* ln1_b     = (const float*)d_in[8];
  const float* in_proj_w = (const float*)d_in[9];
  const float* in_proj_b = (const float*)d_in[10];
  const float* out_proj_w= (const float*)d_in[11];
  const float* out_proj_b= (const float*)d_in[12];
  const float* ln2_s     = (const float*)d_in[13];
  const float* ln2_b     = (const float*)d_in[14];
  const float* ln3_s     = (const float*)d_in[15];
  const float* ln3_b     = (const float*)d_in[16];
  const float* d1_w      = (const float*)d_in[17];
  const float* d1_b      = (const float*)d_in[18];
  const float* ln4_s     = (const float*)d_in[19];
  const float* ln4_b     = (const float*)d_in[20];
  const float* d2_w      = (const float*)d_in[21];
  const float* d2_b      = (const float*)d_in[22];
  const float* gat_wl    = (const float*)d_in[23];
  const float* gat_bl    = (const float*)d_in[24];
  const float* gat_wr    = (const float*)d_in[25];
  const float* gat_br    = (const float*)d_in[26];
  const float* gat_att   = (const float*)d_in[27];
  const float* gat_bias  = (const float*)d_in[28];
  const float* coord_w   = (const float*)d_in[29];
  const float* coord_b   = (const float*)d_in[30];

  float* ws    = (float*)d_ws;
  float* h     = ws + OFF_H;     // residual / hidden (N x 512)
  float* hn    = ws + OFF_HN;    // ln output, reused as padded feat (N x 576)
  float* qkv   = ws + OFF_QKV;   // N x 1536
  float* sc    = ws + OFF_SC;    // 4 x 1024 x 1024
  float* obuf  = ws + OFF_O;     // N x 512 (attn out, then d1 out)
  float* xl    = ws + OFF_XL;    // N x 3072 (overlays qkv/sc/obuf)
  float* xr    = ws + OFF_XR;
  float* gout  = ws + OFF_GOUT;
  float* logit = ws + OFF_LOG;
  float* earr  = ws + OFF_EA;
  float* mx    = ws + OFF_MX;
  float* z     = ws + OFF_Z;
  float* coord = ws + OFF_CO;

  const float inv_sqrt_h = 0.044194173824159216f; // 1/sqrt(512)

  // --- input stage ---
  k_init_coord<<<(NN + 255) / 256, 256, 0, stream>>>(x, coord);
  k_lin_selu<<<(NN * HH) / 256, 256, 0, stream>>>(x, local_feat, conv_feat,
                                                  lin_w, lin_b, h);

  // --- transformer block ---
  k_layernorm<<<NN, 256, 0, stream>>>(h, ln1_s, ln1_b, nullptr, hn);
  // qkv = hn @ in_proj_w^T + b   (in_proj_w: 1536 x 512 row-major -> transB)
  launch_gemm(stream, hn, HH, 0, in_proj_w, HH, 0, true, in_proj_b, nullptr,
              qkv, 3 * HH, 0, NN, 3 * HH, HH, HH, 1.f, 0, 1);
  // scores = q @ k^T / sqrt(H), batched over 4
  launch_gemm(stream, qkv, 3 * HH, (long long)NPMc * 3 * HH,
              qkv + HH, 3 * HH, (long long)NPMc * 3 * HH, true,
              nullptr, nullptr, sc, NPMc, (long long)NPMc * NPMc,
              NPMc, NPMc, HH, HH, inv_sqrt_h, 0, BB);
  k_softmax<<<BB * NPMc, 256, 0, stream>>>(sc, NPMc);
  // o_pre = attn @ v
  launch_gemm(stream, sc, NPMc, (long long)NPMc * NPMc,
              qkv + 2 * HH, 3 * HH, (long long)NPMc * 3 * HH, false,
              nullptr, nullptr, obuf, HH, (long long)NPMc * HH,
              NPMc, HH, NPMc, NPMc, 1.f, 0, BB);
  // o = o_pre @ out_proj_w^T + b  -> hn (reuse)
  launch_gemm(stream, obuf, HH, 0, out_proj_w, HH, 0, true, out_proj_b, nullptr,
              hn, HH, 0, NN, HH, HH, HH, 1.f, 0, 1);
  // h = ln2(o) + residual(h)
  k_layernorm<<<NN, 256, 0, stream>>>(hn, ln2_s, ln2_b, h, h);
  // MLP
  k_layernorm<<<NN, 256, 0, stream>>>(h, ln3_s, ln3_b, nullptr, hn);
  launch_gemm(stream, hn, HH, 0, d1_w, HH, 0, false, d1_b, nullptr,
              obuf, HH, 0, NN, HH, HH, HH, 1.f, /*gelu*/1, 1);
  k_layernorm<<<NN, 256, 0, stream>>>(obuf, ln4_s, ln4_b, nullptr, hn);
  launch_gemm(stream, hn, HH, 0, d2_w, HH, 0, false, d2_b, /*resid*/h,
              h, HH, 0, NN, HH, HH, HH, 1.f, 0, 1);

  // --- GAT loop (hidden lives in h) ---
  for (int it = 0; it < NLOOP; ++it) {
    k_build_feat<<<(NN * FPAD + 255) / 256, 256, 0, stream>>>(coord, h, hn);
    launch_gemm(stream, hn, FPAD, 0, gat_wl, GATW, 0, false, gat_bl, nullptr,
                xl, GATW, 0, NN, GATW, FPAD, FDIM, 1.f, 0, 1);
    launch_gemm(stream, hn, FPAD, 0, gat_wr, GATW, 0, false, gat_br, nullptr,
                xr, GATW, 0, NN, GATW, FPAD, FDIM, 1.f, 0, 1);
    k_fill<<<(unsigned)((SZ_MZ + 255) / 256), 256, 0, stream>>>(mx, -INFINITY, SZ_MZ);
    k_fill<<<(unsigned)((SZ_MZ + 255) / 256), 256, 0, stream>>>(z, 0.f, SZ_MZ);
    k_fill<<<(unsigned)((SZ_XG + 255) / 256), 256, 0, stream>>>(gout, 0.f, SZ_XG);
    k_edge_logits<<<(EE * NHEAD * 32 + 255) / 256, 256, 0, stream>>>(
        edge_index, xl, xr, gat_att, logit, mx);
    k_edge_exp<<<(EE * NHEAD + 255) / 256, 256, 0, stream>>>(
        edge_index, logit, mx, earr, z);
    k_edge_scatter<<<EE, 256, 0, stream>>>(edge_index, earr, z, xl, gout);
    k_gat_hidden<<<(NN * HH) / 256, 256, 0, stream>>>(gout, gat_bias, h);
    k_coord_update<<<(NN * 32) / 256, 256, 0, stream>>>(h, coord_w, coord_b, coord);
  }

  k_copy<<<(2 * NN + 255) / 256, 256, 0, stream>>>(coord, (float*)d_out, 2 * NN);
}
